// MultiHeadSelfAttention_249108103766
// MI455X (gfx1250) — matrix operand order repaired and checked by reading, not compiled, not run on hardware
//
#include <hip/hip_runtime.h>
#include <hip/hip_bf16.h>

// ---------------------------------------------------------------------------
// MI455X (gfx1250) multi-head self-attention, bf16 WMMA pipeline.
// Compute-bound (~137 GFLOP vs <200MB HBM): maximize v_wmma_f32_16x16x32_bf16
// density; packed 16B ds_store_b128 fragment staging; GEMM software-pipelines
// global->register loads under the WMMA phase; attention prefetches next tile.
// ---------------------------------------------------------------------------

typedef __attribute__((ext_vector_type(16))) __bf16 v16bf;
typedef __attribute__((ext_vector_type(8)))  float  v8f;

#define EMBED  1024
#define HEADS  16
#define HDIM   64
#define SEQ    2048
#define BATCH  4
#define ROWS   (BATCH * SEQ)      // 8192
#define ATT_SCALE 0.125f          // 64^-0.5

__device__ __forceinline__ unsigned short f2bf(float f) {
    union { float f; unsigned int u; } v; v.f = f;
    unsigned int u = v.u;
    unsigned int r = u + 0x7FFFu + ((u >> 16) & 1u);   // round-to-nearest-even
    return (unsigned short)(r >> 16);
}
__device__ __forceinline__ unsigned int pk2(float a, float b) {
    return (unsigned int)f2bf(a) | ((unsigned int)f2bf(b) << 16);
}

// ---------------------------------------------------------------------------
// GEMM: C[M,N] = A[M,K] * W[K,N] + bias.
// Block 256 thr (8 waves as 4Mx2N), tile 128x128, BK=64.
// Wave tile 32x64: 16 WMMAs per K-iteration.
// Software pipeline: tile k+1 global loads land in registers while tile k's
// WMMAs run from LDS; registers are flushed to LDS at the next iteration top.
// ---------------------------------------------------------------------------
#define BM 128
#define BN 128
#define BK 64

template <bool ABF16, bool OUTBF16>
__global__ __launch_bounds__(256)
void gemm_bias_wmma(const void* __restrict__ Avoid,
                    const float* __restrict__ W,
                    const float* __restrict__ bias,
                    void* __restrict__ Cvoid,
                    int M, int N, int K)
{
    __shared__ unsigned short sA[(BM / 16) * (BK / 32) * 512];   // 16 KB
    __shared__ unsigned short sB[(BN / 16) * (BK / 32) * 512];   // 16 KB

    const int tid    = threadIdx.x;
    const int lane   = tid & 31;
    const int wave   = tid >> 5;
    const int half   = lane >> 4;
    const int ln     = lane & 15;
    const int wm     = wave >> 1;           // 0..3  (M sub-block, 32 rows)
    const int wn     = wave & 1;            // 0..1  (N sub-block, 64 cols)
    const int blockM = blockIdx.y * BM;
    const int blockN = blockIdx.x * BN;

    // staging indices
    const int am  = tid >> 1;               // A row in tile (0..127)
    const int ak0 = (tid & 1) * 32;         // A k base (0/32)
    const int bn  = tid & 127;              // B col in tile (0..127)
    const int bk0 = (tid >> 7) * 32;        // B k base (0/32)

    unsigned int rA[4][4];                  // staged A regs (next tile)
    unsigned int rB[2][8];                  // staged B regs (next tile)

    auto gload = [&](int k0) {
        if (ABF16) {
            const uint4* s4 = (const uint4*)((const unsigned short*)Avoid +
                              (size_t)(blockM + am) * K + k0 + ak0);
            #pragma unroll
            for (int g = 0; g < 4; ++g) {
                uint4 q = s4[g];
                rA[g][0] = q.x; rA[g][1] = q.y; rA[g][2] = q.z; rA[g][3] = q.w;
            }
        } else {
            const float4* s4 = (const float4*)((const float*)Avoid +
                               (size_t)(blockM + am) * K + k0 + ak0);
            #pragma unroll
            for (int g = 0; g < 4; ++g) {
                float4 fa = s4[2 * g], fb = s4[2 * g + 1];
                rA[g][0] = pk2(fa.x, fa.y); rA[g][1] = pk2(fa.z, fa.w);
                rA[g][2] = pk2(fb.x, fb.y); rA[g][3] = pk2(fb.z, fb.w);
            }
        }
        #pragma unroll
        for (int g16 = 0; g16 < 2; ++g16) {
            int k16 = bk0 + 16 * g16;
            #pragma unroll
            for (int j = 0; j < 8; ++j) {
                float w0 = W[(size_t)(k0 + k16 + 2 * j) * N + blockN + bn];
                float w1 = W[(size_t)(k0 + k16 + 2 * j + 1) * N + blockN + bn];
                rB[g16][j] = pk2(w0, w1);
            }
        }
    };

    v8f acc[2][4] = {};

    gload(0);
    for (int k0 = 0; k0 < K; k0 += BK) {
        // ---- flush staged registers to LDS in fragment order
        #pragma unroll
        for (int g = 0; g < 4; ++g) {
            int k8 = ak0 + 8 * g;
            int kk = k8 & 31;
            int pos = (((am >> 4) * 2 + (k8 >> 5)) * 512) +
                      (((am & 15) + 16 * (kk >> 4)) * 16) + (kk & 15);
            uint4 st; st.x = rA[g][0]; st.y = rA[g][1]; st.z = rA[g][2]; st.w = rA[g][3];
            *(uint4*)(sA + pos) = st;
        }
        #pragma unroll
        for (int g16 = 0; g16 < 2; ++g16) {
            int k16 = bk0 + 16 * g16;
            int pos = (((bn >> 4) * 2 + (k16 >> 5)) * 512) +
                      (((bn & 15) + 16 * ((k16 >> 4) & 1)) * 16);
            uint4 s0; s0.x = rB[g16][0]; s0.y = rB[g16][1]; s0.z = rB[g16][2]; s0.w = rB[g16][3];
            uint4 s1; s1.x = rB[g16][4]; s1.y = rB[g16][5]; s1.z = rB[g16][6]; s1.w = rB[g16][7];
            *(uint4*)(sB + pos)     = s0;
            *(uint4*)(sB + pos + 8) = s1;
        }
        __syncthreads();

        // ---- issue next tile's global loads; they complete under the WMMAs
        if (k0 + BK < K) gload(k0 + BK);

        #pragma unroll
        for (int s = 0; s < 2; ++s) {
            v16bf a0 = *(const v16bf*)(const void*)(sA + ((wm * 2 + 0) * 2 + s) * 512 + lane * 16);
            v16bf a1 = *(const v16bf*)(const void*)(sA + ((wm * 2 + 1) * 2 + s) * 512 + lane * 16);
            #pragma unroll
            for (int t = 0; t < 4; ++t) {
                v16bf bf = *(const v16bf*)(const void*)(sB + ((wn * 4 + t) * 2 + s) * 512 + lane * 16);
                acc[0][t] = __builtin_amdgcn_wmma_f32_16x16x32_bf16(
                    false, a0, false, bf, (short)0, acc[0][t], false, false);
                acc[1][t] = __builtin_amdgcn_wmma_f32_16x16x32_bf16(
                    false, a1, false, bf, (short)0, acc[1][t], false, false);
            }
        }
        __syncthreads();
    }

    // ---- epilogue: bias, store
    #pragma unroll
    for (int i = 0; i < 2; ++i) {
        #pragma unroll
        for (int t = 0; t < 4; ++t) {
            const int col = blockN + wn * 64 + t * 16 + ln;
            const float bv = bias[col];
            #pragma unroll
            for (int r = 0; r < 8; ++r) {
                const int row = blockM + wm * 32 + i * 16 + r + 8 * half;
                const float v = acc[i][t][r] + bv;
                if (OUTBF16)
                    ((unsigned short*)Cvoid)[(size_t)row * N + col] = f2bf(v);
                else
                    ((float*)Cvoid)[(size_t)row * N + col] = v;
            }
        }
    }
}

// ---------------------------------------------------------------------------
// Flash attention: block = (b, h, 64-row Q tile), 4 waves; each wave owns
// 16 Q rows.  Key tiles of 64: S = Q*K^T (8 WMMA), online softmax (max via
// shfl_xor; row-sum via a "ones-column" V fragment so l accumulates on the
// matrix unit), P*V (8 WMMA + 2 ones-WMMA).  Next K/V tile is prefetched.
// ---------------------------------------------------------------------------
__global__ __launch_bounds__(128)
void attn_flash_wmma(const unsigned short* __restrict__ Q,
                     const unsigned short* __restrict__ Kt,
                     const unsigned short* __restrict__ Vt,
                     unsigned short* __restrict__ O)
{
    __shared__ unsigned short sQ[8 * 512];   // (wave, s)
    __shared__ unsigned short sK[8 * 512];   // (key16 j, s=d32)
    __shared__ unsigned short sV[8 * 512];   // (d16 t, s=key32)
    __shared__ unsigned short sP[8 * 512];   // (wave, s=col32)

    const int tid  = threadIdx.x;
    const int lane = tid & 31;
    const int wave = tid >> 5;
    const int half = lane >> 4;
    const int ln   = lane & 15;

    const int qt = blockIdx.x & 31;
    const int h  = (blockIdx.x >> 5) & 15;
    const int b  = blockIdx.x >> 9;
    const int qbase = qt * 64;
    const size_t rowbase = (size_t)(b * SEQ) * EMBED + h * HDIM;

    // ---- stage Q tile (64x64 bf16): 4x 16B loads -> 4x 16B LDS stores
    {
        const int m  = tid >> 1;
        const int d0 = (tid & 1) * 32;
        const uint4* s4 = (const uint4*)(Q + rowbase + (size_t)(qbase + m) * EMBED + d0);
        #pragma unroll
        for (int g = 0; g < 4; ++g) {
            uint4 q = s4[g];
            int k8 = d0 + 8 * g;
            int kk = k8 & 31;
            int pos = (((m >> 4) * 2 + (k8 >> 5)) * 512) +
                      (((m & 15) + 16 * (kk >> 4)) * 16) + (kk & 15);
            *(uint4*)(sQ + pos) = q;
        }
    }
    __syncthreads();

    v16bf qf[2];
    qf[0] = *(const v16bf*)(const void*)(sQ + (wave * 2 + 0) * 512 + lane * 16);
    qf[1] = *(const v16bf*)(const void*)(sQ + (wave * 2 + 1) * 512 + lane * 16);

    // ones-column fragment: B[key][0] = 1.0 for all 64 keys -> lanes ln==0
    v16bf onesf;
    {
        union { unsigned int u[8]; v16bf v; } ob;
        const unsigned int pat = (ln == 0) ? 0x3F803F80u : 0u;
        #pragma unroll
        for (int j = 0; j < 8; ++j) ob.u[j] = pat;
        onesf = ob.v;
    }

    float mrow[8];
    #pragma unroll
    for (int r = 0; r < 8; ++r) mrow[r] = -3.0e38f;
    v8f oacc[4] = {};
    v8f lacc = {};

    for (int kt = 0; kt < SEQ / 64; ++kt) {
        const int keybase = kt * 64;
        // ---- stage K (64x64): row strips, 4x 16B loads -> 4x 16B stores
        {
            const int key = tid >> 1;
            const int d0  = (tid & 1) * 32;
            const uint4* s4 = (const uint4*)(Kt + rowbase + (size_t)(keybase + key) * EMBED + d0);
            #pragma unroll
            for (int g = 0; g < 4; ++g) {
                uint4 q = s4[g];
                int d8 = d0 + 8 * g;
                int kk = d8 & 31;
                int pos = (((key >> 4) * 2 + (d8 >> 5)) * 512) +
                          (((key & 15) + 16 * (kk >> 4)) * 16) + (kk & 15);
                *(uint4*)(sK + pos) = q;
            }
        }
        // ---- stage V (64x64): column strips (fixed d, 32 keys/thread)
        {
            const int d  = tid & 63;
            const int kb = (tid >> 6) * 32;
            #pragma unroll
            for (int g16 = 0; g16 < 2; ++g16) {
                int k16 = kb + 16 * g16;
                const unsigned short* src = Vt + rowbase + (size_t)(keybase + k16) * EMBED + d;
                unsigned int u[8];
                #pragma unroll
                for (int j = 0; j < 8; ++j) {
                    unsigned int lo = src[(size_t)(2 * j) * EMBED];
                    unsigned int hi = src[(size_t)(2 * j + 1) * EMBED];
                    u[j] = lo | (hi << 16);
                }
                int pos = (((d >> 4) * 2 + (k16 >> 5)) * 512) +
                          (((d & 15) + 16 * ((k16 >> 4) & 1)) * 16);
                uint4 s0; s0.x = u[0]; s0.y = u[1]; s0.z = u[2]; s0.w = u[3];
                uint4 s1; s1.x = u[4]; s1.y = u[5]; s1.z = u[6]; s1.w = u[7];
                *(uint4*)(sV + pos)     = s0;
                *(uint4*)(sV + pos + 8) = s1;
            }
        }
        __syncthreads();

        // ---- prefetch next key tile (global_prefetch_b8, no VGPR cost)
        if (kt + 1 < SEQ / 64) {
            const unsigned short* nk = Kt + rowbase +
                (size_t)(keybase + 64 + (tid >> 1)) * EMBED + (tid & 1) * 32;
            const unsigned short* nv = Vt + rowbase +
                (size_t)(keybase + 64 + (tid >> 6) * 32) * EMBED + (tid & 63);
            __builtin_prefetch(nk, 0, 0);
            __builtin_prefetch(nv, 0, 0);
        }

        // ---- S = Q * K^T : 16 rows x 64 keys per wave
        v8f sacc[4] = {};
        #pragma unroll
        for (int s = 0; s < 2; ++s) {
            #pragma unroll
            for (int j = 0; j < 4; ++j) {
                v16bf kf = *(const v16bf*)(const void*)(sK + (j * 2 + s) * 512 + lane * 16);
                sacc[j] = __builtin_amdgcn_wmma_f32_16x16x32_bf16(
                    false, qf[s], false, kf, (short)0, sacc[j], false, false);
            }
        }

        // ---- online softmax (rows at vgpr r + 8*half; cols across 16 lanes)
        #pragma unroll
        for (int r = 0; r < 8; ++r) {
            float x0 = sacc[0][r] * ATT_SCALE;
            float x1 = sacc[1][r] * ATT_SCALE;
            float x2 = sacc[2][r] * ATT_SCALE;
            float x3 = sacc[3][r] * ATT_SCALE;
            float mv = fmaxf(fmaxf(x0, x1), fmaxf(x2, x3));
            #pragma unroll
            for (int off = 1; off <= 8; off <<= 1)
                mv = fmaxf(mv, __shfl_xor(mv, off, 32));
            float mnew = fmaxf(mrow[r], mv);
            float sc   = __expf(mrow[r] - mnew);
            mrow[r] = mnew;
            float p0 = __expf(x0 - mnew);
            float p1 = __expf(x1 - mnew);
            float p2 = __expf(x2 - mnew);
            float p3 = __expf(x3 - mnew);
            lacc[r] *= sc;
            #pragma unroll
            for (int t = 0; t < 4; ++t) oacc[t][r] *= sc;
            // P -> LDS in A-frag order (s = col/32, lane half = (col/16)&1, e = col&15)
            int lane_a = (r + 8 * half);
            unsigned short* pw0 = sP + (wave * 2 + 0) * 512 + lane_a * 16;
            unsigned short* pw1 = sP + (wave * 2 + 1) * 512 + lane_a * 16;
            int eb = ln;
            pw0[eb]       = f2bf(p0);
            pw0[eb + 256] = f2bf(p1);
            pw1[eb]       = f2bf(p2);
            pw1[eb + 256] = f2bf(p3);
        }

        // ---- O += P * V ; l accumulates through the ones-column fragment
        #pragma unroll
        for (int s = 0; s < 2; ++s) {
            v16bf pf = *(const v16bf*)(const void*)(sP + (wave * 2 + s) * 512 + lane * 16);
            #pragma unroll
            for (int t = 0; t < 4; ++t) {
                v16bf vf = *(const v16bf*)(const void*)(sV + (t * 2 + s) * 512 + lane * 16);
                oacc[t] = __builtin_amdgcn_wmma_f32_16x16x32_bf16(
                    false, pf, false, vf, (short)0, oacc[t], false, false);
            }
            lacc = __builtin_amdgcn_wmma_f32_16x16x32_bf16(
                false, pf, false, onesf, (short)0, lacc, false, false);
        }
        __syncthreads();
    }

    // ---- l lives in column 0 (lanes 0/16): broadcast, normalize, store
    float inv[8];
    #pragma unroll
    for (int r = 0; r < 8; ++r)
        inv[r] = 1.0f / __shfl(lacc[r], 16 * half, 32);
    #pragma unroll
    for (int t = 0; t < 4; ++t) {
        #pragma unroll
        for (int r = 0; r < 8; ++r) {
            int row = qbase + wave * 16 + r + 8 * half;
            O[rowbase + (size_t)row * EMBED + t * 16 + ln] = f2bf(oacc[t][r] * inv[r]);
        }
    }
}

// ---------------------------------------------------------------------------
extern "C" void kernel_launch(void* const* d_in, const int* in_sizes, int n_in,
                              void* d_out, int out_size, void* d_ws, size_t ws_size,
                              hipStream_t stream) {
    (void)in_sizes; (void)n_in; (void)out_size; (void)ws_size;
    const float* x   = (const float*)d_in[0];
    const float* w_q = (const float*)d_in[1];
    const float* b_q = (const float*)d_in[2];
    const float* w_k = (const float*)d_in[3];
    const float* b_k = (const float*)d_in[4];
    const float* w_v = (const float*)d_in[5];
    const float* b_v = (const float*)d_in[6];
    const float* w_o = (const float*)d_in[7];
    const float* b_o = (const float*)d_in[8];

    const size_t BUF = (size_t)ROWS * EMBED;      // elements per bf16 buffer
    unsigned short* qb = (unsigned short*)d_ws;
    unsigned short* kb = qb + BUF;
    unsigned short* vb = kb + BUF;
    unsigned short* ob = vb + BUF;

    dim3 ggrid(EMBED / BN, ROWS / BM);            // (8, 64)
    gemm_bias_wmma<false, true><<<ggrid, 256, 0, stream>>>(x, w_q, b_q, qb, ROWS, EMBED, EMBED);
    gemm_bias_wmma<false, true><<<ggrid, 256, 0, stream>>>(x, w_k, b_k, kb, ROWS, EMBED, EMBED);
    gemm_bias_wmma<false, true><<<ggrid, 256, 0, stream>>>(x, w_v, b_v, vb, ROWS, EMBED, EMBED);

    attn_flash_wmma<<<BATCH * HEADS * (SEQ / 64), 128, 0, stream>>>(qb, kb, vb, ob);

    gemm_bias_wmma<true, false><<<ggrid, 256, 0, stream>>>(ob, w_o, b_o, (float*)d_out, ROWS, EMBED, EMBED);
}
